// vgg_ssd_73246372266025
// MI455X (gfx1250) — compile-verified, matrix operand order checked by reading, not run on hardware
//
#include <hip/hip_runtime.h>
#include <hip/hip_fp16.h>
#include <math.h>
#include <stdint.h>

typedef __attribute__((ext_vector_type(16))) _Float16 v16h;
typedef __attribute__((ext_vector_type(8)))  _Float16 v8h;
typedef __attribute__((ext_vector_type(8)))  float    v8f;

// ---------------------------------------------------------------------------
// elementwise f32 -> f16
// ---------------------------------------------------------------------------
__global__ void k_f32_to_f16(const float* __restrict__ src, _Float16* __restrict__ dst, int n) {
    int i = blockIdx.x * blockDim.x + threadIdx.x;
    if (i < n) dst[i] = (_Float16)src[i];
}

// ---------------------------------------------------------------------------
// Repack conv weights OIHW(f32) -> [CoutP][K*K][CinP](f16), zero padded.
// GEMM K-dim ordering becomes (kh,kw,ci) so each 32-wide K step stays inside
// one (kh,kw) plane (CinP is a multiple of 32).
// ---------------------------------------------------------------------------
__global__ void k_pack_w(const float* __restrict__ w, _Float16* __restrict__ dst,
                         int Cout, int Cin, int KK, int CinP, long long total)
{
    long long idx = (long long)blockIdx.x * blockDim.x + threadIdx.x;
    if (idx >= total) return;
    int ci  = (int)(idx % CinP);
    long long t = idx / CinP;
    int khw = (int)(t % KK);
    int row = (int)(t / KK);
    float v = 0.0f;
    if (row < Cout && ci < Cin) v = w[((size_t)row * Cin + ci) * KK + khw];
    dst[idx] = (_Float16)v;
}

// ---------------------------------------------------------------------------
// Implicit-GEMM conv, f16 in / f16 out, f32 accumulate via WMMA.
// Block = 256 threads (8 waves). Block tile: 64 Cout x 256 pixels.
// Wave tile: 64 Cout x 32 pixels  -> 4 M-subtiles x 2 N-subtiles = 8 WMMA/step.
// A (weight) 64x32 f16 tile is staged to LDS with async loads, double buffered.
// ---------------------------------------------------------------------------
__global__ __launch_bounds__(256) void k_conv_wmma(
    const _Float16* __restrict__ in, const _Float16* __restrict__ wp,
    const float* __restrict__ bias, _Float16* __restrict__ out,
    int N, int Cin, int CinP, int Hin, int Win, int Cout, int Hout, int Wout,
    int K, int pad, int stride, int dil, int relu)
{
    __shared__ __align__(16) _Float16 ldsA[2][64 * 32];

    const int tid   = threadIdx.x;
    const int lane  = tid & 31;
    const int wave  = tid >> 5;
    const int col   = lane & 15;
    const int kHalf = (lane & 16) ? 8 : 0;    // CDNA5 16-bit A/B K split per lane-half

    const int KK  = K * K;
    const int KgP = KK * CinP;
    const int HW  = Hin * Win;
    const int Ng  = N * Hout * Wout;

    const int coutBase = blockIdx.y << 6;
    const int pixWave  = (blockIdx.x << 8) + (wave << 5);

    // per-N-subtile pixel decode (constant for whole kernel)
    bool pok[2]; int n_[2], oh_[2], ow_[2], ihB[2], iwB[2]; size_t img[2];
    #pragma unroll
    for (int ni = 0; ni < 2; ++ni) {
        int pix = pixWave + (ni << 4) + col;
        pok[ni] = pix < Ng;
        int pp = pok[ni] ? pix : 0;
        ow_[ni] = pp % Wout; pp /= Wout;
        oh_[ni] = pp % Hout; n_[ni] = pp / Hout;
        ihB[ni] = oh_[ni] * stride - pad;
        iwB[ni] = ow_[ni] * stride - pad;
        img[ni] = (size_t)n_[ni] * Cin * HW;
    }

    // async staging of the 64x32 A tile: one b128 per thread (4KB total)
    const unsigned ldsB0 = (unsigned)(uintptr_t)&ldsA[0][0];
    const unsigned ldsB1 = (unsigned)(uintptr_t)&ldsA[1][0];
    const int rowS = tid >> 2;
    const int chS  = (tid & 3) << 3;
    auto stage = [&](int bf, int kOff) {
        unsigned lo = (bf ? ldsB1 : ldsB0) + (unsigned)tid * 16u;
        unsigned long long ga = (unsigned long long)(uintptr_t)
            (wp + (size_t)(coutBase + rowS) * KgP + kOff + chS);
        asm volatile("global_load_async_to_lds_b128 %0, %1, off"
                     :: "v"(lo), "v"(ga) : "memory");
    };

    v8f acc[4][2];
    #pragma unroll
    for (int mi = 0; mi < 4; ++mi)
        #pragma unroll
        for (int ni = 0; ni < 2; ++ni)
            acc[mi][ni] = (v8f){0.f,0.f,0.f,0.f,0.f,0.f,0.f,0.f};

    stage(0, 0);
    asm volatile("s_wait_asynccnt 0" ::: "memory");
    __syncthreads();

    int s = 0;
    for (int khw = 0; khw < KK; ++khw) {
        const int kh = khw / K, kw = khw % K;
        // hoisted per-plane spatial offsets / bounds
        int sp[2]; bool ok[2];
        #pragma unroll
        for (int ni = 0; ni < 2; ++ni) {
            int ih = ihB[ni] + kh * dil;
            int iw = iwB[ni] + kw * dil;
            ok[ni] = pok[ni] && (unsigned)ih < (unsigned)Hin && (unsigned)iw < (unsigned)Win;
            sp[ni] = ih * Win + iw;
        }
        for (int cib = 0; cib < CinP; cib += 32, ++s) {
            const int bf = s & 1;
            // prefetch next K-step while computing this one
            {
                int cib2 = cib + 32, khw2 = khw;
                if (cib2 == CinP) { cib2 = 0; khw2++; }
                if (khw2 < KK) stage(bf ^ 1, khw2 * CinP + cib2);
            }
            // A fragments: two ds_load_b128 per M-subtile
            v16h afr[4];
            #pragma unroll
            for (int mi = 0; mi < 4; ++mi) {
                const int rl = (mi << 4) + col;
                v8h lo = *(const v8h*)&ldsA[bf][rl * 32 + kHalf];
                v8h hi = *(const v8h*)&ldsA[bf][rl * 32 + 16 + kHalf];
                afr[mi] = __builtin_shufflevector(lo, hi,
                    0,1,2,3,4,5,6,7,8,9,10,11,12,13,14,15);
            }
            // B fragments: strided f16 gather, no divides in the loop
            v16h bfr[2];
            #pragma unroll
            for (int ni = 0; ni < 2; ++ni) {
                v16h b = {};
                if (ok[ni]) {
                    const _Float16* bp = in + img[ni] + sp[ni];
                    #pragma unroll
                    for (int e = 0; e < 16; ++e) {
                        int ciE = cib + ((e < 8) ? (kHalf + e) : (8 + kHalf + e));
                        b[e] = (ciE < Cin) ? bp[(size_t)ciE * HW] : (_Float16)0.0f;
                    }
                }
                bfr[ni] = b;
            }
            // 8 WMMAs, EXEC all-1s here (uniform control flow)
            #pragma unroll
            for (int mi = 0; mi < 4; ++mi)
                #pragma unroll
                for (int ni = 0; ni < 2; ++ni)
                    acc[mi][ni] = __builtin_amdgcn_wmma_f32_16x16x32_f16(
                        false, afr[mi], false, bfr[ni], (short)0, acc[mi][ni], false, false);

            asm volatile("s_wait_asynccnt 0" ::: "memory");
            __syncthreads();
        }
    }

    // epilogue: bias + ReLU + f16 store
    #pragma unroll
    for (int mi = 0; mi < 4; ++mi) {
        #pragma unroll
        for (int r = 0; r < 8; ++r) {
            const int co = coutBase + (mi << 4) + r + ((lane & 16) ? 8 : 0);
            if (co < Cout) {
                const float bsv = bias[co];
                #pragma unroll
                for (int ni = 0; ni < 2; ++ni) {
                    if (pok[ni]) {
                        float v = acc[mi][ni][r] + bsv;
                        if (relu) v = fmaxf(v, 0.0f);
                        out[(((size_t)n_[ni] * Cout + co) * Hout + oh_[ni]) * Wout + ow_[ni]]
                            = (_Float16)v;
                    }
                }
            }
        }
    }
}

// ---------------------------------------------------------------------------
// Max pooling on f16 activations
// ---------------------------------------------------------------------------
__global__ void k_maxpool16(const _Float16* __restrict__ in, _Float16* __restrict__ out,
                            int N, int C, int Hin, int Win, int Hout, int Wout,
                            int k, int s, int padLo)
{
    int idx = blockIdx.x * blockDim.x + threadIdx.x;
    int total = N * C * Hout * Wout;
    if (idx >= total) return;
    int ow = idx % Wout; int t = idx / Wout;
    int oh = t % Hout;   t /= Hout;
    int c  = t % C;      int n = t / C;
    const _Float16* base = in + ((size_t)n * C + c) * Hin * Win;
    float m = -INFINITY;
    for (int kh = 0; kh < k; ++kh) {
        int ih = oh * s - padLo + kh;
        if ((unsigned)ih >= (unsigned)Hin) continue;
        for (int kw = 0; kw < k; ++kw) {
            int iw = ow * s - padLo + kw;
            if ((unsigned)iw >= (unsigned)Win) continue;
            m = fmaxf(m, (float)base[(size_t)ih * Win + iw]);
        }
    }
    out[idx] = (_Float16)m;
}

// ---------------------------------------------------------------------------
// L2Norm across channels (relu4_3)
// ---------------------------------------------------------------------------
__global__ void k_l2norm16(const _Float16* __restrict__ in, const float* __restrict__ w,
                           _Float16* __restrict__ out, int N, int C, int HW)
{
    int idx = blockIdx.x * blockDim.x + threadIdx.x;
    if (idx >= N * HW) return;
    int n = idx / HW, p = idx % HW;
    const _Float16* base = in + (size_t)n * C * HW + p;
    float ss = 0.0f;
    for (int c = 0; c < C; ++c) { float v = (float)base[(size_t)c * HW]; ss += v * v; }
    float inv = 1.0f / (sqrtf(ss) + 1e-10f);
    _Float16* ob = out + (size_t)n * C * HW + p;
    for (int c = 0; c < C; ++c)
        ob[(size_t)c * HW] = (_Float16)((float)base[(size_t)c * HW] * inv * w[c]);
}

// ---------------------------------------------------------------------------
// Head output: NCHW f16 -> per-image [prior_off + (h*W+w)*C + c] f32
// ---------------------------------------------------------------------------
__global__ void k_head_permute(const _Float16* __restrict__ in, float* __restrict__ dst,
                               int N, int C, int H, int W, int imgStride, int baseOff)
{
    int idx = blockIdx.x * blockDim.x + threadIdx.x;
    int total = N * C * H * W;
    if (idx >= total) return;
    int w = idx % W; int t = idx / W;
    int h = t % H;   t /= H;
    int c = t % C;   int n = t / C;
    dst[(size_t)n * imgStride + baseOff + ((size_t)h * W + w) * C + c] = (float)in[idx];
}

// ---------------------------------------------------------------------------
// softmax over 21 classes, in place
// ---------------------------------------------------------------------------
__global__ void k_softmax21(float* __restrict__ conf, int rows)
{
    int r = blockIdx.x * blockDim.x + threadIdx.x;
    if (r >= rows) return;
    float* p = conf + (size_t)r * 21;
    float mx = p[0];
    for (int c = 1; c < 21; ++c) mx = fmaxf(mx, p[c]);
    float e[21], sum = 0.0f;
    for (int c = 0; c < 21; ++c) { e[c] = expf(p[c] - mx); sum += e[c]; }
    float inv = 1.0f / sum;
    for (int c = 0; c < 21; ++c) p[c] = e[c] * inv;
}

// ---------------------------------------------------------------------------
// SSD box decode
// ---------------------------------------------------------------------------
__global__ void k_decode(const float* __restrict__ loc, const float* __restrict__ priors,
                         float* __restrict__ boxes, int total)
{
    int idx = blockIdx.x * blockDim.x + threadIdx.x;
    if (idx >= total) return;
    int p = idx % 8732;
    const float* l  = loc    + (size_t)idx * 4;
    const float* pr = priors + (size_t)p * 4;
    float cx = pr[0] + l[0] * 0.1f * pr[2];
    float cy = pr[1] + l[1] * 0.1f * pr[3];
    float w  = pr[2] * expf(l[2] * 0.2f);
    float h  = pr[3] * expf(l[3] * 0.2f);
    float* o = boxes + (size_t)idx * 4;
    o[0] = cx - 0.5f * w;  o[1] = cy - 0.5f * h;
    o[2] = cx + 0.5f * w;  o[3] = cy + 0.5f * h;
}

__global__ void k_fill0(float* __restrict__ p, int n)
{
    int i = blockIdx.x * blockDim.x + threadIdx.x;
    if (i < n) p[i] = 0.0f;
}

// ---------------------------------------------------------------------------
// Per-(image,class) NMS, one workgroup each; scores staged in LDS.
// ---------------------------------------------------------------------------
#define NMS_TOPK 200
__global__ __launch_bounds__(256) void k_nms(const float* __restrict__ conf,
                                             const float* __restrict__ boxes,
                                             float* __restrict__ out)
{
    __shared__ float sc[8732];
    __shared__ float rv[256];
    __shared__ int   ri[256];
    __shared__ float ts[NMS_TOPK];
    __shared__ int   tidx[NMS_TOPK];
    __shared__ float bb[NMS_TOPK][4];
    __shared__ int   keep[NMS_TOPK];

    const int b   = blockIdx.x / 20;
    const int cls = 1 + blockIdx.x % 20;
    const int tid = threadIdx.x;
    const float* cf = conf  + (size_t)b * 8732 * 21;
    const float* bx = boxes + (size_t)b * 8732 * 4;

    for (int i = tid; i < 8732; i += 256) {
        float s = cf[(size_t)i * 21 + cls];
        sc[i] = (s > 0.01f) ? s : 0.0f;
    }
    __syncthreads();

    for (int r = 0; r < NMS_TOPK; ++r) {
        float best = -2.0f; int bi = 0;
        for (int i = tid; i < 8732; i += 256) {
            float v = sc[i];
            if (v > best) { best = v; bi = i; }
        }
        rv[tid] = best; ri[tid] = bi;
        __syncthreads();
        for (int off = 128; off > 0; off >>= 1) {
            if (tid < off && rv[tid + off] > rv[tid]) {
                rv[tid] = rv[tid + off]; ri[tid] = ri[tid + off];
            }
            __syncthreads();
        }
        if (tid == 0) { ts[r] = rv[0]; tidx[r] = ri[0]; sc[ri[0]] = -1.0f; }
        __syncthreads();
    }

    for (int r = tid; r < NMS_TOPK; r += 256) {
        int p = tidx[r];
        bb[r][0] = bx[(size_t)p * 4 + 0];
        bb[r][1] = bx[(size_t)p * 4 + 1];
        bb[r][2] = bx[(size_t)p * 4 + 2];
        bb[r][3] = bx[(size_t)p * 4 + 3];
        keep[r] = (ts[r] > 0.01f) ? 1 : 0;
    }
    __syncthreads();

    for (int i = 0; i < NMS_TOPK; ++i) {
        __syncthreads();
        if (!keep[i]) continue;                        // uniform
        const float ax1 = bb[i][0], ay1 = bb[i][1], ax2 = bb[i][2], ay2 = bb[i][3];
        const float ai  = (ax2 - ax1) * (ay2 - ay1);
        for (int j = i + 1 + tid; j < NMS_TOPK; j += 256) {
            if (!keep[j]) continue;
            float ix1 = fmaxf(ax1, bb[j][0]);
            float iy1 = fmaxf(ay1, bb[j][1]);
            float ix2 = fminf(ax2, bb[j][2]);
            float iy2 = fminf(ay2, bb[j][3]);
            float iw  = fmaxf(ix2 - ix1, 0.0f);
            float ih  = fmaxf(iy2 - iy1, 0.0f);
            float inter = iw * ih;
            float aj = (bb[j][2] - bb[j][0]) * (bb[j][3] - bb[j][1]);
            float iou = inter / fmaxf(ai + aj - inter, 1e-12f);
            if (iou > 0.45f) keep[j] = 0;
        }
    }
    __syncthreads();

    if (tid == 0) {
        int pos = 0;
        for (int i = 0; i < NMS_TOPK; ++i) {
            if (!keep[i]) continue;
            float* o = out + (((size_t)b * 21 + cls) * NMS_TOPK + pos) * 5;
            o[0] = ts[i]; o[1] = bb[i][0]; o[2] = bb[i][1]; o[3] = bb[i][2]; o[4] = bb[i][3];
            ++pos;
        }
    }
}

// ---------------------------------------------------------------------------
// Host orchestration
// ---------------------------------------------------------------------------
extern "C" void kernel_launch(void* const* d_in, const int* in_sizes, int n_in,
                              void* d_out, int out_size, void* d_ws, size_t ws_size,
                              hipStream_t stream)
{
    (void)in_sizes; (void)n_in; (void)out_size; (void)ws_size;
    const int B = 4;

    const float* x = (const float*)d_in[0];
    auto VW = [&](int i) { return (const float*)d_in[1 + 2 * i]; };
    auto VB = [&](int i) { return (const float*)d_in[2 + 2 * i]; };
    auto EW = [&](int i) { return (const float*)d_in[31 + 2 * i]; };
    auto EB = [&](int i) { return (const float*)d_in[32 + 2 * i]; };
    auto LW = [&](int i) { return (const float*)d_in[47 + 2 * i]; };
    auto LB = [&](int i) { return (const float*)d_in[48 + 2 * i]; };
    auto CW = [&](int i) { return (const float*)d_in[59 + 2 * i]; };
    auto CB = [&](int i) { return (const float*)d_in[60 + 2 * i]; };
    const float* l2w    = (const float*)d_in[71];
    const float* priors = (const float*)d_in[72];

    // ---- workspace carving (bytes, 256B aligned) ----
    char* base = (char*)d_ws;
    size_t off = 0;
    auto carve = [&](size_t bytes) {
        void* p = base + off; off += (bytes + 255) & ~(size_t)255; return p;
    };
    const size_t ACT = (size_t)B * 64 * 300 * 300;           // max activation elems
    _Float16* actA = (_Float16*)carve(ACT * 2);
    _Float16* actB = (_Float16*)carve(ACT * 2);
    _Float16* xh   = (_Float16*)carve((size_t)B * 3 * 300 * 300 * 2);
    _Float16* w16p = (_Float16*)carve((size_t)4718592 * 2);  // 1024*9*512 halves max
    _Float16* s0 = (_Float16*)carve((size_t)B * 512 * 38 * 38 * 2);
    _Float16* s1 = (_Float16*)carve((size_t)B * 1024 * 19 * 19 * 2);
    _Float16* s2 = (_Float16*)carve((size_t)B * 512 * 10 * 10 * 2);
    _Float16* s3 = (_Float16*)carve((size_t)B * 256 * 5 * 5 * 2);
    _Float16* s4 = (_Float16*)carve((size_t)B * 256 * 3 * 3 * 2);
    _Float16* s5 = (_Float16*)carve((size_t)B * 256 * 1 * 1 * 2);
    float* locb  = (float*)carve((size_t)B * 8732 * 4 * 4);
    float* confb = (float*)carve((size_t)B * 8732 * 21 * 4);
    float* boxes = (float*)carve((size_t)B * 8732 * 4 * 4);

    // ---- helpers ----
    auto conv = [&](const _Float16* in, const float* w, const float* bias, _Float16* out,
                    int Cin, int Hin, int Win, int Cout, int K, int pad, int st, int dil,
                    int relu, int& Hout, int& Wout) {
        Hout = (Hin + 2 * pad - dil * (K - 1) - 1) / st + 1;
        Wout = (Win + 2 * pad - dil * (K - 1) - 1) / st + 1;
        const int KK = K * K;
        const int CinP  = (Cin + 31) & ~31;
        const int CoutP = (Cout + 63) & ~63;
        long long np = (long long)CoutP * KK * CinP;
        k_pack_w<<<(int)((np + 255) / 256), 256, 0, stream>>>(w, w16p, Cout, Cin, KK, CinP, np);
        dim3 grid((B * Hout * Wout + 255) / 256, CoutP / 64);
        k_conv_wmma<<<grid, 256, 0, stream>>>(in, w16p, bias, out, B, Cin, CinP,
                                              Hin, Win, Cout, Hout, Wout, K, pad, st, dil, relu);
    };
    auto pool = [&](const _Float16* in, _Float16* out, int C, int Hin, int Win,
                    int k, int s, int pLo, int pHi, int& Hout, int& Wout) {
        Hout = (Hin + pLo + pHi - k) / s + 1;
        Wout = (Win + pLo + pHi - k) / s + 1;
        int total = B * C * Hout * Wout;
        k_maxpool16<<<(total + 255) / 256, 256, 0, stream>>>(in, out, B, C, Hin, Win,
                                                             Hout, Wout, k, s, pLo);
    };

    // ---- input to f16 ----
    {
        int n = B * 3 * 300 * 300;
        k_f32_to_f16<<<(n + 255) / 256, 256, 0, stream>>>(x, xh, n);
    }

    int Ho = 300, Wo = 300;
    // ---- VGG backbone ----
    conv(xh,   VW(0),  VB(0),  actA, 3,    Ho, Wo, 64,   3, 1, 1, 1, 1, Ho, Wo);
    conv(actA, VW(1),  VB(1),  actB, 64,   Ho, Wo, 64,   3, 1, 1, 1, 1, Ho, Wo);
    pool(actB, actA, 64, Ho, Wo, 2, 2, 0, 0, Ho, Wo);                     // 150
    conv(actA, VW(2),  VB(2),  actB, 64,   Ho, Wo, 128,  3, 1, 1, 1, 1, Ho, Wo);
    conv(actB, VW(3),  VB(3),  actA, 128,  Ho, Wo, 128,  3, 1, 1, 1, 1, Ho, Wo);
    pool(actA, actB, 128, Ho, Wo, 2, 2, 0, 0, Ho, Wo);                    // 75
    conv(actB, VW(4),  VB(4),  actA, 128,  Ho, Wo, 256,  3, 1, 1, 1, 1, Ho, Wo);
    conv(actA, VW(5),  VB(5),  actB, 256,  Ho, Wo, 256,  3, 1, 1, 1, 1, Ho, Wo);
    conv(actB, VW(6),  VB(6),  actA, 256,  Ho, Wo, 256,  3, 1, 1, 1, 1, Ho, Wo);
    pool(actA, actB, 256, Ho, Wo, 2, 2, 0, 1, Ho, Wo);                    // ceil -> 38
    conv(actB, VW(7),  VB(7),  actA, 256,  Ho, Wo, 512,  3, 1, 1, 1, 1, Ho, Wo);
    conv(actA, VW(8),  VB(8),  actB, 512,  Ho, Wo, 512,  3, 1, 1, 1, 1, Ho, Wo);
    conv(actB, VW(9),  VB(9),  actA, 512,  Ho, Wo, 512,  3, 1, 1, 1, 1, Ho, Wo); // relu4_3
    {
        int total = B * 38 * 38;
        k_l2norm16<<<(total + 255) / 256, 256, 0, stream>>>(actA, l2w, s0, B, 512, 38 * 38);
    }
    pool(actA, actB, 512, Ho, Wo, 2, 2, 0, 0, Ho, Wo);                    // 19
    conv(actB, VW(10), VB(10), actA, 512,  Ho, Wo, 512,  3, 1, 1, 1, 1, Ho, Wo);
    conv(actA, VW(11), VB(11), actB, 512,  Ho, Wo, 512,  3, 1, 1, 1, 1, Ho, Wo);
    conv(actB, VW(12), VB(12), actA, 512,  Ho, Wo, 512,  3, 1, 1, 1, 1, Ho, Wo);
    pool(actA, actB, 512, Ho, Wo, 3, 1, 1, 1, Ho, Wo);                    // pool5 -> 19
    conv(actB, VW(13), VB(13), actA, 512,  Ho, Wo, 1024, 3, 6, 1, 6, 1, Ho, Wo); // conv6 dil6
    conv(actA, VW(14), VB(14), s1,   1024, Ho, Wo, 1024, 1, 0, 1, 1, 1, Ho, Wo); // conv7

    // ---- extras ----
    conv(s1,   EW(0), EB(0), actA, 1024, 19, 19, 256, 1, 0, 1, 1, 1, Ho, Wo);
    conv(actA, EW(1), EB(1), s2,   256,  Ho, Wo, 512, 3, 1, 2, 1, 1, Ho, Wo);    // 10
    conv(s2,   EW(2), EB(2), actA, 512,  Ho, Wo, 128, 1, 0, 1, 1, 1, Ho, Wo);
    conv(actA, EW(3), EB(3), s3,   128,  Ho, Wo, 256, 3, 1, 2, 1, 1, Ho, Wo);    // 5
    conv(s3,   EW(4), EB(4), actA, 256,  Ho, Wo, 128, 1, 0, 1, 1, 1, Ho, Wo);
    conv(actA, EW(5), EB(5), s4,   128,  Ho, Wo, 256, 3, 0, 1, 1, 1, Ho, Wo);    // 3
    conv(s4,   EW(6), EB(6), actA, 256,  Ho, Wo, 128, 1, 0, 1, 1, 1, Ho, Wo);
    conv(actA, EW(7), EB(7), s5,   128,  Ho, Wo, 256, 3, 0, 1, 1, 1, Ho, Wo);    // 1

    // ---- multibox heads ----
    const _Float16* srcs[6] = { s0, s1, s2, s3, s4, s5 };
    const int sch[6]  = { 512, 1024, 512, 256, 256, 256 };
    const int shw[6]  = { 38, 19, 10, 5, 3, 1 };
    const int mbox[6] = { 4, 6, 6, 6, 4, 4 };
    const int poff[6] = { 0, 5776, 7942, 8542, 8692, 8728 };

    for (int sI = 0; sI < 6; ++sI) {
        int Cout, Hh, Wh, total;
        Cout = mbox[sI] * 4;
        conv(srcs[sI], LW(sI), LB(sI), actB, sch[sI], shw[sI], shw[sI], Cout, 3, 1, 1, 1, 0, Hh, Wh);
        total = B * Cout * Hh * Wh;
        k_head_permute<<<(total + 255) / 256, 256, 0, stream>>>(
            actB, locb, B, Cout, Hh, Wh, 8732 * 4, poff[sI] * 4);
        Cout = mbox[sI] * 21;
        conv(srcs[sI], CW(sI), CB(sI), actB, sch[sI], shw[sI], shw[sI], Cout, 3, 1, 1, 1, 0, Hh, Wh);
        total = B * Cout * Hh * Wh;
        k_head_permute<<<(total + 255) / 256, 256, 0, stream>>>(
            actB, confb, B, Cout, Hh, Wh, 8732 * 21, poff[sI] * 21);
    }

    // ---- softmax, decode, NMS ----
    {
        int rows = B * 8732;
        k_softmax21<<<(rows + 255) / 256, 256, 0, stream>>>(confb, rows);
        k_decode<<<(rows + 255) / 256, 256, 0, stream>>>(locb, priors, boxes, rows);
    }
    {
        int tot = B * 21 * NMS_TOPK * 5;
        k_fill0<<<(tot + 255) / 256, 256, 0, stream>>>((float*)d_out, tot);
    }
    k_nms<<<B * 20, 256, 0, stream>>>(confb, boxes, (float*)d_out);
}
